// SelfAttention_16174846837382
// MI455X (gfx1250) — compile-verified, matrix operand order checked by reading, not run on hardware
//
#include <hip/hip_runtime.h>
#include <hip/hip_bf16.h>

typedef __bf16 bf16_t;
typedef __bf16 v16bf  __attribute__((ext_vector_type(16)));
typedef __bf16 bf16x8 __attribute__((ext_vector_type(8)));
typedef __bf16 bf16x4 __attribute__((ext_vector_type(4)));
typedef float  v8f    __attribute__((ext_vector_type(8)));
typedef float  f32x4  __attribute__((ext_vector_type(4)));

#define WMMA_BF16(a, b, c) \
  __builtin_amdgcn_wmma_f32_16x16x32_bf16(false, (a), false, (b), (short)0, (c), false, false)

constexpr int B_   = 2;
constexpr int S_   = 2048;
constexpr int EMB_ = 1024;
constexpr int H_   = 16;
constexpr int DH_  = 64;
constexpr int MT_  = B_ * S_;  // 4096 total rows

// ---------------------------------------------------------------------------
// Fragment loader: 16x32 bf16 tile from row-major storage (row stride `ld`
// elements) into the CDNA5 WMMA A/B operand layout:
//   lanes 0-15 : row = lane,    K = [0..7] and [16..23]
//   lanes 16-31: row = lane-16, K = [8..15] and [24..31]
// Two 16-byte loads per lane. Works for global and LDS (generic) pointers.
// Loading a row-major [N,K] matrix this way yields the B operand of A*B^T.
// ---------------------------------------------------------------------------
__device__ __forceinline__ v16bf load_frag(const bf16_t* p0, int ld) {
  const int lane = threadIdx.x & 31;
  const bf16_t* p = p0 + (size_t)(lane & 15) * ld + ((lane >> 4) << 3);
  bf16x8 lo = *(const bf16x8*)(p);
  bf16x8 hi = *(const bf16x8*)(p + 16);
  v16bf f;
#pragma unroll
  for (int i = 0; i < 8; ++i) { f[i] = lo[i]; f[i + 8] = hi[i]; }
  return f;
}

__device__ __forceinline__ float half_reduce_max(float v) {
  v = fmaxf(v, __shfl_xor(v, 1, 32));
  v = fmaxf(v, __shfl_xor(v, 2, 32));
  v = fmaxf(v, __shfl_xor(v, 4, 32));
  v = fmaxf(v, __shfl_xor(v, 8, 32));
  return v;
}
__device__ __forceinline__ float half_reduce_sum(float v) {
  v += __shfl_xor(v, 1, 32);
  v += __shfl_xor(v, 2, 32);
  v += __shfl_xor(v, 4, 32);
  v += __shfl_xor(v, 8, 32);
  return v;
}

// ---------------------------------------------------------------------------
// fp32 -> bf16 conversion (vectorized x4)
// ---------------------------------------------------------------------------
__global__ void __launch_bounds__(256)
cvt_f32_to_bf16(const float* __restrict__ x, bf16_t* __restrict__ y, int n4) {
  int i = blockIdx.x * blockDim.x + threadIdx.x;
  if (i >= n4) return;
  f32x4 v = ((const f32x4*)x)[i];
  bf16x4 o;
  o.x = (bf16_t)v.x; o.y = (bf16_t)v.y; o.z = (bf16_t)v.z; o.w = (bf16_t)v.w;
  ((bf16x4*)y)[i] = o;
}

// ---------------------------------------------------------------------------
// Fused QKV GEMM: qkv[m,e] = sum_k X[m,k]*W[e,k] + bias[e]
// M=4096, N=3072, K=1024. Wave -> 32x64 tile (2x4 WMMA accs, 1.5 b128
// loads per WMMA); block = 8 waves (2M x 4N) = 64 x 256. Scatter epilogue
// writes per-head Q[bh,s,d], K[bh,s,d] and transposed Vt[bh,d,s] (bf16).
// ---------------------------------------------------------------------------
__global__ void __launch_bounds__(256)
qkv_gemm(const bf16_t* __restrict__ X, const bf16_t* __restrict__ W,
         const float* __restrict__ bias,
         bf16_t* __restrict__ Qb, bf16_t* __restrict__ Kb, bf16_t* __restrict__ Vt) {
  const int wave = threadIdx.x >> 5, lane = threadIdx.x & 31;
  const int m0 = blockIdx.y * 64 + (wave >> 2) * 32;
  const int n0 = blockIdx.x * 256 + (wave & 3) * 64;

  const v8f z = {0.f, 0.f, 0.f, 0.f, 0.f, 0.f, 0.f, 0.f};
  v8f acc[2][4] = {{z, z, z, z}, {z, z, z, z}};

  for (int k = 0; k < EMB_; k += 32) {
    v16bf a0 = load_frag(X + (size_t)m0 * EMB_ + k, EMB_);
    v16bf a1 = load_frag(X + (size_t)(m0 + 16) * EMB_ + k, EMB_);
#pragma unroll
    for (int j = 0; j < 4; ++j) {
      v16bf b = load_frag(W + (size_t)(n0 + j * 16) * EMB_ + k, EMB_);
      acc[0][j] = WMMA_BF16(a0, b, acc[0][j]);
      acc[1][j] = WMMA_BF16(a1, b, acc[1][j]);
    }
  }

  const int cn = lane & 15, mb = (lane >> 4) * 8;
#pragma unroll
  for (int i = 0; i < 2; ++i)
#pragma unroll
    for (int j = 0; j < 4; ++j)
#pragma unroll
      for (int r = 0; r < 8; ++r) {
        int m = m0 + i * 16 + mb + r;
        int e = n0 + j * 16 + cn;
        bf16_t bv = (bf16_t)(acc[i][j][r] + bias[e]);
        int b = m >> 11, s = m & (S_ - 1);
        int region = e >> 10, hd = e & 1023, h = hd >> 6, d = hd & 63;
        size_t bh = (size_t)(b * H_ + h);
        if (region == 0)      Qb[(bh * S_ + s) * DH_ + d] = bv;
        else if (region == 1) Kb[(bh * S_ + s) * DH_ + d] = bv;
        else                  Vt[(bh * DH_ + d) * S_ + s] = bv;
      }
}

// ---------------------------------------------------------------------------
// Flash attention: one wave per 16-query tile of one (b,h); 8 waves per
// block walk the same head's K/V stream (WGP$ reuse).
// 32-key blocks: QK^T (4 WMMAs) -> online softmax (f32, shfl reductions)
// -> P through LDS (C-layout -> A-layout transpose) -> P*V (4 WMMAs).
// Output written bf16 as [b, s, h*64+d] = [b, s, 1024] for the final GEMM.
// ---------------------------------------------------------------------------
__global__ void __launch_bounds__(256)
attn_kernel(const bf16_t* __restrict__ Q, const bf16_t* __restrict__ K,
            const bf16_t* __restrict__ Vt, bf16_t* __restrict__ O) {
  const int wave = threadIdx.x >> 5, lane = threadIdx.x & 31;
  const int bh = blockIdx.y;
  const int q0 = (blockIdx.x * 8 + wave) * 16;

  const bf16_t* Qh = Q + (size_t)bh * S_ * DH_;
  const bf16_t* Kh = K + (size_t)bh * S_ * DH_;
  const bf16_t* Vh = Vt + (size_t)bh * DH_ * S_;

  // 16 rows x (32 + 8 pad) halves per wave: rows 80B apart -> 16B aligned,
  // distinct banks for the b128 fragment reads.
  __shared__ alignas(16) bf16_t pshm[8][16 * 40];
  bf16_t* pw = &pshm[wave][0];

  v16bf qf0 = load_frag(Qh + (size_t)q0 * DH_ + 0, DH_);
  v16bf qf1 = load_frag(Qh + (size_t)q0 * DH_ + 32, DH_);

  const v8f z = {0.f, 0.f, 0.f, 0.f, 0.f, 0.f, 0.f, 0.f};
  v8f oacc[4] = {z, z, z, z};
  float mrow[8], lrow[8];
#pragma unroll
  for (int r = 0; r < 8; ++r) { mrow[r] = -3.0e38f; lrow[r] = 0.f; }

  const int cn = lane & 15, mb = (lane >> 4) * 8;
  const float sc = 0.125f;  // 1/sqrt(64)

  for (int kb = 0; kb < S_; kb += 32) {
    v16bf k00 = load_frag(Kh + (size_t)kb * DH_ + 0, DH_);
    v16bf k01 = load_frag(Kh + (size_t)kb * DH_ + 32, DH_);
    v16bf k10 = load_frag(Kh + (size_t)(kb + 16) * DH_ + 0, DH_);
    v16bf k11 = load_frag(Kh + (size_t)(kb + 16) * DH_ + 32, DH_);
    v8f s0 = WMMA_BF16(qf1, k01, WMMA_BF16(qf0, k00, z));
    v8f s1 = WMMA_BF16(qf1, k11, WMMA_BF16(qf0, k10, z));

    float corr[8];
#pragma unroll
    for (int r = 0; r < 8; ++r) {
      float a = s0[r] * sc, b = s1[r] * sc;
      float mx = half_reduce_max(fmaxf(a, b));
      float mnew = fmaxf(mrow[r], mx);
      float pa = __expf(a - mnew), pb = __expf(b - mnew);
      float ps = half_reduce_sum(pa + pb);
      float c = __expf(mrow[r] - mnew);
      lrow[r] = lrow[r] * c + ps;
      mrow[r] = mnew;
      corr[r] = c;
      s0[r] = pa; s1[r] = pb;
    }
#pragma unroll
    for (int t = 0; t < 4; ++t)
#pragma unroll
      for (int r = 0; r < 8; ++r) oacc[t][r] *= corr[r];

    // C-layout (lane = key col, vgpr = row) -> row-major 16x32 bf16 in LDS
#pragma unroll
    for (int r = 0; r < 8; ++r) {
      pw[(mb + r) * 40 + cn]      = (bf16_t)s0[r];
      pw[(mb + r) * 40 + 16 + cn] = (bf16_t)s1[r];
    }
    asm volatile("s_wait_dscnt 0x0" ::: "memory");
    v16bf pf = load_frag(pw, 40);  // A operand: P 16(q) x 32(key)

    v16bf vf0 = load_frag(Vh + (size_t)0  * S_ + kb, S_);
    v16bf vf1 = load_frag(Vh + (size_t)16 * S_ + kb, S_);
    v16bf vf2 = load_frag(Vh + (size_t)32 * S_ + kb, S_);
    v16bf vf3 = load_frag(Vh + (size_t)48 * S_ + kb, S_);
    oacc[0] = WMMA_BF16(pf, vf0, oacc[0]);
    oacc[1] = WMMA_BF16(pf, vf1, oacc[1]);
    oacc[2] = WMMA_BF16(pf, vf2, oacc[2]);
    oacc[3] = WMMA_BF16(pf, vf3, oacc[3]);
  }

  const int h = bh & 15, b = bh >> 4;
#pragma unroll
  for (int t = 0; t < 4; ++t)
#pragma unroll
    for (int r = 0; r < 8; ++r) {
      int s = q0 + mb + r;
      int d = t * 16 + cn;
      O[((size_t)b * S_ + s) * EMB_ + h * DH_ + d] = (bf16_t)(oacc[t][r] / lrow[r]);
    }
}

// ---------------------------------------------------------------------------
// Output projection: out[m,e] = sum_k A[m,k]*W[e,k] + bias[e]  (fp32 out)
// M=4096, N=1024, K=1024. Same 32x64-per-wave tiling as qkv_gemm.
// ---------------------------------------------------------------------------
__global__ void __launch_bounds__(256)
proj_gemm(const bf16_t* __restrict__ A, const bf16_t* __restrict__ W,
          const float* __restrict__ bias, float* __restrict__ out) {
  const int wave = threadIdx.x >> 5, lane = threadIdx.x & 31;
  const int m0 = blockIdx.y * 64 + (wave >> 2) * 32;
  const int n0 = blockIdx.x * 256 + (wave & 3) * 64;

  const v8f z = {0.f, 0.f, 0.f, 0.f, 0.f, 0.f, 0.f, 0.f};
  v8f acc[2][4] = {{z, z, z, z}, {z, z, z, z}};

  for (int k = 0; k < EMB_; k += 32) {
    v16bf a0 = load_frag(A + (size_t)m0 * EMB_ + k, EMB_);
    v16bf a1 = load_frag(A + (size_t)(m0 + 16) * EMB_ + k, EMB_);
#pragma unroll
    for (int j = 0; j < 4; ++j) {
      v16bf b = load_frag(W + (size_t)(n0 + j * 16) * EMB_ + k, EMB_);
      acc[0][j] = WMMA_BF16(a0, b, acc[0][j]);
      acc[1][j] = WMMA_BF16(a1, b, acc[1][j]);
    }
  }

  const int cn = lane & 15, mb = (lane >> 4) * 8;
#pragma unroll
  for (int i = 0; i < 2; ++i)
#pragma unroll
    for (int j = 0; j < 4; ++j)
#pragma unroll
      for (int r = 0; r < 8; ++r) {
        int m = m0 + i * 16 + mb + r;
        int e = n0 + j * 16 + cn;
        out[(size_t)m * EMB_ + e] = acc[i][j][r] + bias[e];
      }
}

// ---------------------------------------------------------------------------
extern "C" void kernel_launch(void* const* d_in, const int* in_sizes, int n_in,
                              void* d_out, int out_size, void* d_ws, size_t ws_size,
                              hipStream_t stream) {
  (void)in_sizes; (void)n_in; (void)out_size; (void)ws_size;
  const float* x     = (const float*)d_in[0];
  const float* w_in  = (const float*)d_in[1];
  const float* b_in  = (const float*)d_in[2];
  const float* w_out = (const float*)d_in[3];
  const float* b_out = (const float*)d_in[4];
  float* out = (float*)d_out;

  // workspace carve-up (bf16), all 16B-aligned sizes
  bf16_t* p = (bf16_t*)d_ws;
  bf16_t* xb  = p; p += (size_t)MT_ * EMB_;       // 4M
  bf16_t* wib = p; p += (size_t)3 * EMB_ * EMB_;  // 3M
  bf16_t* wob = p; p += (size_t)EMB_ * EMB_;      // 1M
  bf16_t* Qb  = p; p += (size_t)MT_ * EMB_;       // 4M  [bh,s,d]
  bf16_t* Kb  = p; p += (size_t)MT_ * EMB_;       // 4M  [bh,s,d]
  bf16_t* Vt  = p; p += (size_t)MT_ * EMB_;       // 4M  [bh,d,s]
  bf16_t* Ob  = p; p += (size_t)MT_ * EMB_;       // 4M  [b,s,e]

  int n4;
  n4 = MT_ * EMB_ / 4;
  cvt_f32_to_bf16<<<(n4 + 255) / 256, 256, 0, stream>>>(x, xb, n4);
  n4 = 3 * EMB_ * EMB_ / 4;
  cvt_f32_to_bf16<<<(n4 + 255) / 256, 256, 0, stream>>>(w_in, wib, n4);
  n4 = EMB_ * EMB_ / 4;
  cvt_f32_to_bf16<<<(n4 + 255) / 256, 256, 0, stream>>>(w_out, wob, n4);

  qkv_gemm<<<dim3(3 * EMB_ / 256, MT_ / 64), 256, 0, stream>>>(xb, wib, b_in, Qb, Kb, Vt);
  attn_kernel<<<dim3(S_ / 128, B_ * H_), 256, 0, stream>>>(Qb, Kb, Vt, Ob);
  proj_gemm<<<dim3(EMB_ / 256, MT_ / 64), 256, 0, stream>>>(Ob, wob, b_out, out);
}
